// _SGATMLPLite_11123965296939
// MI455X (gfx1250) — matrix operand order repaired and checked by reading, not compiled, not run on hardware
//
#include <hip/hip_runtime.h>
#include <hip/hip_bf16.h>
#include <math.h>

// ---- problem constants (from reference) ----
#define NFEAT   128   // NF
#define DTOK    64    // DT
#define TDIM    16    // TD
#define KSEL    8     // K (top-k)
#define SEL_H   16
#define FFN_HID 32
#define COR_HID 32
#define FUS_HID 16
#define GDIM    80    // DT + TD

typedef __attribute__((ext_vector_type(16))) _Float16 v16h;
typedef __attribute__((ext_vector_type(8)))  float    v8f;

union AFrag { v16h v; _Float16 h[16]; };
union CFrag { v8f  v; float    f[8];  };

struct Params {
    const float* x_num;   // (B, NF)
    const float* W_num;   // (NF, DT)
    const float* b_num;   // (NF, DT)
    const float* cls_tok; // (DT,)
    const float* W_base;  // (DT, 1)
    const float* b_base;  // (1,)
    const float* ln_s_g;  // (DT,)
    const float* ln_s_b;  // (DT,)
    const float* W_s1;    // (DT, SEL_H)
    const float* b_s1;    // (SEL_H,)
    const float* W_s2;    // (SEL_H, 1)
    const float* b_s2;    // (1,)
    const float* W_p;     // (DT, TD)
    const float* b_p;     // (TD,)
    const float* ln1_g, *ln1_b;           // (TD,)
    const float* Wq, *bq, *Wk, *bk, *Wv, *bv, *Wo, *bo; // (TD,TD)/(TD,)
    const float* ln2_g, *ln2_b;           // (TD,)
    const float* W_f1, *b_f1;             // (TD,FFN_HID)/(FFN_HID,)
    const float* W_f2, *b_f2;             // (FFN_HID,TD)/(TD,)
    const float* ln_c_g, *ln_c_b;         // (TD,)
    const float* W_c1, *b_c1;             // (TD,COR_HID)/(COR_HID,)
    const float* W_c2, *b_c2;             // (COR_HID,1)/(1,)
    const float* ln_f_g, *ln_f_b;         // (GDIM,)
    const float* W_g1, *b_g1;             // (GDIM,FUS_HID)/(FUS_HID,)
    const float* W_g2, *b_g2;             // (FUS_HID,1)/(1,)
};

__device__ __forceinline__ float fast_rcp(float x) { return __builtin_amdgcn_rcpf(x); }

// branch-free tanh: tanh(x) = 1 - 2/(e^{2x}+1), |x| clamped so e^{2x} stays finite.
__device__ __forceinline__ float fast_tanh(float x) {
    float xc = fminf(fmaxf(x, -15.f), 15.f);
    float e  = __expf(2.f * xc);                  // v_exp_f32
    return fmaf(-2.f, fast_rcp(e + 1.f), 1.f);    // v_rcp_f32
}
__device__ __forceinline__ float gelu_tanh(float x) {
    const float c0 = 0.7978845608028654f; // sqrt(2/pi)
    return 0.5f * x * (1.0f + fast_tanh(c0 * (x + 0.044715f * x * x * x)));
}
__device__ __forceinline__ float fast_sigmoid(float x) {
    return fast_rcp(1.f + __expf(-x));
}

__device__ __forceinline__ float wave_sum(float v) {
    v += __shfl_xor(v, 16, 32);
    v += __shfl_xor(v, 8, 32);
    v += __shfl_xor(v, 4, 32);
    v += __shfl_xor(v, 2, 32);
    v += __shfl_xor(v, 1, 32);
    return v;
}

__global__ __launch_bounds__(32)
void sgat_fused_kernel(Params p, float* __restrict__ out) {
    const int lane = threadIdx.x;   // wave32: one wave per row
    const int m    = lane & 15;     // WMMA row (A) / col (B,C)
    const int half = lane >> 4;
    const int b    = blockIdx.x;

    __shared__ float s_scores[NFEAT];
    __shared__ float s_lnsg[DTOK], s_lnsb[DTOK];
    __shared__ float s_part[16][20];      // score partials transpose tile (pad 20: 16B rows, no conflicts)
    __shared__ float s_topv[KSEL];
    __shared__ int   s_topi[KSEL];
    __shared__ float s_t [KSEL][TDIM];
    __shared__ float s_xn[KSEL][TDIM];
    __shared__ float s_q [KSEL][TDIM];
    __shared__ float s_k [KSEL][TDIM];
    __shared__ float s_v [KSEL][TDIM];
    __shared__ float s_att[KSEL][KSEL];
    __shared__ float s_hid[KSEL][FFN_HID];
    __shared__ float s_sum[TDIM];
    __shared__ float s_misc[2];

    __builtin_prefetch(p.x_num + (size_t)b * NFEAT, 0, 1); // global_prefetch_b8

    // stage scorer LN params in LDS (reused 8x per wave, read back as b128)
    for (int d = lane; d < DTOK; d += 32) {
        s_lnsg[d] = p.ln_s_g[d];
        s_lnsb[d] = p.ln_s_b[d];
    }
    __syncthreads();

    // ---- constant B fragments: W_s1 (64x16) and W_p (64x16) in f16 ----
    // B layout: lane n = lane&15; element j: K=chunk*32+(j/8)*16+half*8+(j%8)
    AFrag bs[2], bp[2];
    const int n = m;
#pragma unroll
    for (int c = 0; c < 2; ++c)
#pragma unroll
        for (int j = 0; j < 16; ++j) {
            int k = c * 32 + (j >> 3) * 16 + half * 8 + (j & 7);
            bs[c].h[j] = (_Float16)p.W_s1[k * SEL_H + n];
            bp[c].h[j] = (_Float16)p.W_p [k * TDIM  + n];
        }

    const float bs1n = p.b_s1[n];
    const float ws2n = p.W_s2[n];
    const float bs2  = p.b_s2[0];

    // ================= Stage 1: scorer over 128 tokens, 8 WMMA tiles =================
    // A-fragment column map (ISA 7.12.2, 16-bit A 16x32): for chunk c, vector group g,
    // element i: col = c*32 + g*16 + half*8 + i  -> 8 consecutive floats per group.
    for (int tt = 0; tt < NFEAT / 16; ++tt) {
        const int f = tt * 16 + m;
        const float x = p.x_num[(size_t)b * NFEAT + f];
        const float* wrow = p.W_num + (size_t)f * DTOK;
        const float* brow = p.b_num + (size_t)f * DTOK;

        float vraw[2][16];
        float sum = 0.f, ssq = 0.f;
#pragma unroll
        for (int c = 0; c < 2; ++c)
#pragma unroll
            for (int g = 0; g < 2; ++g) {
                const int base = c * 32 + g * 16 + half * 8;
                const float4 w0 = *reinterpret_cast<const float4*>(wrow + base);
                const float4 w1 = *reinterpret_cast<const float4*>(wrow + base + 4);
                const float4 c0 = *reinterpret_cast<const float4*>(brow + base);
                const float4 c1 = *reinterpret_cast<const float4*>(brow + base + 4);
                float* vp = &vraw[c][g * 8];
                vp[0] = fmaf(x, w0.x, c0.x); vp[1] = fmaf(x, w0.y, c0.y);
                vp[2] = fmaf(x, w0.z, c0.z); vp[3] = fmaf(x, w0.w, c0.w);
                vp[4] = fmaf(x, w1.x, c1.x); vp[5] = fmaf(x, w1.y, c1.y);
                vp[6] = fmaf(x, w1.z, c1.z); vp[7] = fmaf(x, w1.w, c1.w);
#pragma unroll
                for (int i = 0; i < 8; ++i) { sum += vp[i]; ssq += vp[i] * vp[i]; }
            }
        // token row split across lane pair (l, l^16): combine for LN stats
        sum += __shfl_xor(sum, 16, 32);
        ssq += __shfl_xor(ssq, 16, 32);
        const float mean = sum * (1.f / DTOK);
        const float var  = ssq * (1.f / DTOK) - mean * mean;
        const float rstd = rsqrtf(var + 1e-5f);

        AFrag a[2];
#pragma unroll
        for (int c = 0; c < 2; ++c)
#pragma unroll
            for (int g = 0; g < 2; ++g) {
                const int base = c * 32 + g * 16 + half * 8;
                const float4 g0 = *reinterpret_cast<const float4*>(&s_lnsg[base]);
                const float4 g1 = *reinterpret_cast<const float4*>(&s_lnsg[base + 4]);
                const float4 b0 = *reinterpret_cast<const float4*>(&s_lnsb[base]);
                const float4 b1 = *reinterpret_cast<const float4*>(&s_lnsb[base + 4]);
                const float* vp = &vraw[c][g * 8];
                const float gg[8] = {g0.x, g0.y, g0.z, g0.w, g1.x, g1.y, g1.z, g1.w};
                const float bb[8] = {b0.x, b0.y, b0.z, b0.w, b1.x, b1.y, b1.z, b1.w};
#pragma unroll
                for (int i = 0; i < 8; ++i) {
                    float t = (vp[i] - mean) * rstd;
                    a[c].h[g * 8 + i] = (_Float16)fmaf(t, gg[i], bb[i]);
                }
            }

        CFrag acc;
#pragma unroll
        for (int r = 0; r < 8; ++r) acc.f[r] = bs1n;
        acc.v = __builtin_amdgcn_wmma_f32_16x16x32_f16(false, a[0].v, false, bs[0].v,
                                                       (short)0, acc.v, false, false);
        acc.v = __builtin_amdgcn_wmma_f32_16x16x32_f16(false, a[1].v, false, bs[1].v,
                                                       (short)0, acc.v, false, false);
        // score[m'] = sum_n gelu(acc[m'][n]) * W_s2[n]: transpose partials through LDS
        // C layout: VGPR r holds (M = half*8 + r, N = lane&15)
#pragma unroll
        for (int r = 0; r < 8; ++r)
            s_part[half * 8 + r][n] = gelu_tanh(acc.f[r]) * ws2n;
        __syncthreads();
        if (lane < 16) {
            const float4* rp = reinterpret_cast<const float4*>(&s_part[lane][0]);
            float4 a0 = rp[0], a1 = rp[1], a2 = rp[2], a3 = rp[3];
            float ssum = ((a0.x + a0.y) + (a0.z + a0.w)) + ((a1.x + a1.y) + (a1.z + a1.w))
                       + ((a2.x + a2.y) + (a2.z + a2.w)) + ((a3.x + a3.y) + (a3.z + a3.w));
            s_scores[tt * 16 + lane] = ssum + bs2;
        }
        __syncthreads();   // protect s_part before next tile's stores
    }

    // ================= Stage 2: top-K=8 (descending, low-index tie break) =================
#pragma unroll 1
    for (int kk = 0; kk < KSEL; ++kk) {
        float bestv = -__builtin_inff(); int besti = NFEAT;
#pragma unroll
        for (int j = 0; j < 4; ++j) {
            int i = lane * 4 + j;
            float sv = s_scores[i];
            if (sv > bestv) { bestv = sv; besti = i; }
        }
#pragma unroll
        for (int off = 16; off >= 1; off >>= 1) {
            float ov = __shfl_xor(bestv, off, 32);
            int   oi = __shfl_xor(besti, off, 32);
            if (ov > bestv || (ov == bestv && oi < besti)) { bestv = ov; besti = oi; }
        }
        if (lane == 0) {
            s_topv[kk] = bestv;
            s_topi[kk] = besti;
            s_scores[besti] = -__builtin_inff();
        }
        __syncthreads();
    }

    // ========== Stage 3: project selected raw tokens, t = sel @ W_p + b_p (WMMA, M padded) ==
    {
        AFrag ap[2];
        float xv = 0.f; int fsel = 0;
        if (m < KSEL) { fsel = s_topi[m]; xv = p.x_num[(size_t)b * NFEAT + fsel]; }
        const float* wrow = p.W_num + (size_t)fsel * DTOK;
        const float* brow = p.b_num + (size_t)fsel * DTOK;
        const float msk = (m < KSEL) ? 1.f : 0.f;   // zero-pad rows 8..15
#pragma unroll
        for (int c = 0; c < 2; ++c)
#pragma unroll
            for (int g = 0; g < 2; ++g) {
                const int base = c * 32 + g * 16 + half * 8;
                const float4 w0 = *reinterpret_cast<const float4*>(wrow + base);
                const float4 w1 = *reinterpret_cast<const float4*>(wrow + base + 4);
                const float4 c0 = *reinterpret_cast<const float4*>(brow + base);
                const float4 c1 = *reinterpret_cast<const float4*>(brow + base + 4);
                const float ww[8] = {w0.x, w0.y, w0.z, w0.w, w1.x, w1.y, w1.z, w1.w};
                const float cc[8] = {c0.x, c0.y, c0.z, c0.w, c1.x, c1.y, c1.z, c1.w};
#pragma unroll
                for (int i = 0; i < 8; ++i)
                    ap[c].h[g * 8 + i] = (_Float16)(msk * fmaf(xv, ww[i], cc[i]));
            }
        CFrag tacc;
        const float bpn = p.b_p[n];
#pragma unroll
        for (int r = 0; r < 8; ++r) tacc.f[r] = bpn;
        tacc.v = __builtin_amdgcn_wmma_f32_16x16x32_f16(false, ap[0].v, false, bp[0].v,
                                                        (short)0, tacc.v, false, false);
        tacc.v = __builtin_amdgcn_wmma_f32_16x16x32_f16(false, ap[1].v, false, bp[1].v,
                                                        (short)0, tacc.v, false, false);
        if (half == 0) { // rows M=0..7 are the real tokens
#pragma unroll
            for (int r = 0; r < 8; ++r) s_t[r][n] = tacc.f[r];
        }
    }
    __syncthreads();

    // ================= Stage 4: tiny transformer block on (8 x 16), VALU+LDS ==========
    // LN1
    if (lane < KSEL) {
        float mu = 0.f;
        for (int d = 0; d < TDIM; ++d) mu += s_t[lane][d];
        mu *= (1.f / TDIM);
        float vv = 0.f;
        for (int d = 0; d < TDIM; ++d) { float dd = s_t[lane][d] - mu; vv += dd * dd; }
        float rs = rsqrtf(vv * (1.f / TDIM) + 1e-5f);
        for (int d = 0; d < TDIM; ++d)
            s_xn[lane][d] = (s_t[lane][d] - mu) * rs * p.ln1_g[d] + p.ln1_b[d];
    }
    __syncthreads();
    // q, k, v
    for (int e = lane; e < KSEL * TDIM; e += 32) {
        int mm = e >> 4, nn = e & 15;
        float aq = p.bq[nn], ak = p.bk[nn], av = p.bv[nn];
        for (int d = 0; d < TDIM; ++d) {
            float xnv = s_xn[mm][d];
            aq = fmaf(xnv, p.Wq[d * TDIM + nn], aq);
            ak = fmaf(xnv, p.Wk[d * TDIM + nn], ak);
            av = fmaf(xnv, p.Wv[d * TDIM + nn], av);
        }
        s_q[mm][nn] = aq; s_k[mm][nn] = ak; s_v[mm][nn] = av;
    }
    __syncthreads();
    // attention logits + score bias (bias read from LDS: no scratch)
    for (int e = lane; e < KSEL * KSEL; e += 32) {
        int qi = e >> 3, kj = e & 7;
        float acc = 0.f;
        for (int d = 0; d < TDIM; ++d) acc = fmaf(s_q[qi][d], s_k[kj][d], acc);
        s_att[qi][kj] = acc * 0.25f + s_topv[kj];   // /sqrt(16)
    }
    __syncthreads();
    // softmax rows
    if (lane < KSEL) {
        float mx = -__builtin_inff();
        for (int j = 0; j < KSEL; ++j) mx = fmaxf(mx, s_att[lane][j]);
        float se = 0.f;
        for (int j = 0; j < KSEL; ++j) { float e2 = __expf(s_att[lane][j] - mx); s_att[lane][j] = e2; se += e2; }
        float inv = fast_rcp(se);
        for (int j = 0; j < KSEL; ++j) s_att[lane][j] *= inv;
    }
    __syncthreads();
    // att @ v  (reuse s_q)
    for (int e = lane; e < KSEL * TDIM; e += 32) {
        int mm = e >> 4, nn = e & 15;
        float acc = 0.f;
        for (int j = 0; j < KSEL; ++j) acc = fmaf(s_att[mm][j], s_v[j][nn], acc);
        s_q[mm][nn] = acc;
    }
    __syncthreads();
    // output proj + residual
    for (int e = lane; e < KSEL * TDIM; e += 32) {
        int mm = e >> 4, nn = e & 15;
        float acc = p.bo[nn];
        for (int d = 0; d < TDIM; ++d) acc = fmaf(s_q[mm][d], p.Wo[d * TDIM + nn], acc);
        s_t[mm][nn] += acc;
    }
    __syncthreads();
    // LN2
    if (lane < KSEL) {
        float mu = 0.f;
        for (int d = 0; d < TDIM; ++d) mu += s_t[lane][d];
        mu *= (1.f / TDIM);
        float vv = 0.f;
        for (int d = 0; d < TDIM; ++d) { float dd = s_t[lane][d] - mu; vv += dd * dd; }
        float rs = rsqrtf(vv * (1.f / TDIM) + 1e-5f);
        for (int d = 0; d < TDIM; ++d)
            s_xn[lane][d] = (s_t[lane][d] - mu) * rs * p.ln2_g[d] + p.ln2_b[d];
    }
    __syncthreads();
    // FFN 16 -> 32 (gelu) -> 16, residual
    for (int e = lane; e < KSEL * FFN_HID; e += 32) {
        int mm = e >> 5, jj = e & 31;
        float acc = p.b_f1[jj];
        for (int d = 0; d < TDIM; ++d) acc = fmaf(s_xn[mm][d], p.W_f1[d * FFN_HID + jj], acc);
        s_hid[mm][jj] = gelu_tanh(acc);
    }
    __syncthreads();
    for (int e = lane; e < KSEL * TDIM; e += 32) {
        int mm = e >> 4, nn = e & 15;
        float acc = p.b_f2[nn];
        for (int j = 0; j < FFN_HID; ++j) acc = fmaf(s_hid[mm][j], p.W_f2[j * TDIM + nn], acc);
        s_t[mm][nn] += acc;
    }
    __syncthreads();
    // summary = mean over tokens
    if (lane < TDIM) {
        float acc = 0.f;
        for (int mm = 0; mm < KSEL; ++mm) acc += s_t[mm][lane];
        s_sum[lane] = acc * (1.f / KSEL);
    }
    __syncthreads();

    // ================= Stage 5: heads ================================================
    // correction: delta = relu(LN(summary) @ W_c1) @ W_c2 + b_c2
    if (lane == 0) {
        float mu = 0.f;
        for (int d = 0; d < TDIM; ++d) mu += s_sum[d];
        mu *= (1.f / TDIM);
        float vv = 0.f;
        for (int d = 0; d < TDIM; ++d) { float dd = s_sum[d] - mu; vv += dd * dd; }
        s_misc[0] = mu;
        s_misc[1] = rsqrtf(vv * (1.f / TDIM) + 1e-5f);
    }
    __syncthreads();
    const float muc = s_misc[0], rsc = s_misc[1];
    float hj = p.b_c1[lane];                 // lane = hidden unit j (COR_HID == 32)
    for (int d = 0; d < TDIM; ++d) {
        float cd = (s_sum[d] - muc) * rsc * p.ln_c_g[d] + p.ln_c_b[d];
        hj = fmaf(cd, p.W_c1[d * COR_HID + lane], hj);
    }
    hj = fmaxf(hj, 0.f) * p.W_c2[lane];
    const float delta = wave_sum(hj) + p.b_c2[0];

    // gate: beta = sigmoid(gelu(LN([cls, summary]) @ W_g1) @ W_g2 + b_g2)
    float gsum = 0.f, gssq = 0.f;
    for (int d = lane; d < GDIM; d += 32) {
        float gv = (d < DTOK) ? p.cls_tok[d] : s_sum[d - DTOK];
        gsum += gv; gssq += gv * gv;
    }
    gsum = wave_sum(gsum); gssq = wave_sum(gssq);
    const float mug = gsum * (1.f / GDIM);
    const float rsg = rsqrtf(gssq * (1.f / GDIM) - mug * mug + 1e-5f);
    float bj = 0.f;
    if (lane < FUS_HID) {
        float acc = p.b_g1[lane];
        for (int d = 0; d < GDIM; ++d) {
            float gv = (d < DTOK) ? p.cls_tok[d] : s_sum[d - DTOK];
            float gn = (gv - mug) * rsg * p.ln_f_g[d] + p.ln_f_b[d];
            acc = fmaf(gn, p.W_g1[d * FUS_HID + lane], acc);
        }
        bj = gelu_tanh(acc) * p.W_g2[lane];
    }
    const float blog = wave_sum(bj) + p.b_g2[0];
    const float beta = fast_sigmoid(blog);

    // y_base = cls_tok @ W_base + b_base  (constant across rows, recomputed cheaply)
    float yb = 0.f;
    for (int d = lane; d < DTOK; d += 32) yb = fmaf(p.cls_tok[d], p.W_base[d], yb);
    yb = wave_sum(yb) + p.b_base[0];

    if (lane == 0) out[b] = yb + beta * delta;
}

extern "C" void kernel_launch(void* const* d_in, const int* in_sizes, int n_in,
                              void* d_out, int out_size, void* d_ws, size_t ws_size,
                              hipStream_t stream) {
    // setup_inputs() flat order: x_num, x_cat, then params dict in insertion order.
    Params p;
    p.x_num   = (const float*)d_in[0];
    // d_in[1] = x_cat (int32)  -- dead code in reference (cat tokens never used)
    p.W_num   = (const float*)d_in[2];
    p.b_num   = (const float*)d_in[3];
    // d_in[4] = cat_emb -- dead code
    p.cls_tok = (const float*)d_in[5];
    p.W_base  = (const float*)d_in[6];
    p.b_base  = (const float*)d_in[7];
    p.ln_s_g  = (const float*)d_in[8];
    p.ln_s_b  = (const float*)d_in[9];
    p.W_s1    = (const float*)d_in[10];
    p.b_s1    = (const float*)d_in[11];
    p.W_s2    = (const float*)d_in[12];
    p.b_s2    = (const float*)d_in[13];
    p.W_p     = (const float*)d_in[14];
    p.b_p     = (const float*)d_in[15];
    p.ln1_g   = (const float*)d_in[16];
    p.ln1_b   = (const float*)d_in[17];
    p.Wq = (const float*)d_in[18]; p.bq = (const float*)d_in[19];
    p.Wk = (const float*)d_in[20]; p.bk = (const float*)d_in[21];
    p.Wv = (const float*)d_in[22]; p.bv = (const float*)d_in[23];
    p.Wo = (const float*)d_in[24]; p.bo = (const float*)d_in[25];
    p.ln2_g = (const float*)d_in[26]; p.ln2_b = (const float*)d_in[27];
    p.W_f1 = (const float*)d_in[28]; p.b_f1 = (const float*)d_in[29];
    p.W_f2 = (const float*)d_in[30]; p.b_f2 = (const float*)d_in[31];
    p.ln_c_g = (const float*)d_in[32]; p.ln_c_b = (const float*)d_in[33];
    p.W_c1 = (const float*)d_in[34]; p.b_c1 = (const float*)d_in[35];
    p.W_c2 = (const float*)d_in[36]; p.b_c2 = (const float*)d_in[37];
    p.ln_f_g = (const float*)d_in[38]; p.ln_f_b = (const float*)d_in[39];
    p.W_g1 = (const float*)d_in[40]; p.b_g1 = (const float*)d_in[41];
    p.W_g2 = (const float*)d_in[42]; p.b_g2 = (const float*)d_in[43];

    const int rows = in_sizes[0] / NFEAT;   // B = 16384
    sgat_fused_kernel<<<rows, 32, 0, stream>>>(p, (float*)d_out);
}
